// GraphMSG_82308753260924
// MI455X (gfx1250) — matrix operand order repaired and checked by reading, not compiled, not run on hardware
//
#include <hip/hip_runtime.h>
#include <hip/hip_bf16.h>

typedef unsigned short u16;
typedef unsigned int   u32;
typedef __attribute__((ext_vector_type(16))) __bf16 v16bf;
typedef __attribute__((ext_vector_type(8)))  float  v8f;
typedef __attribute__((ext_vector_type(4)))  float  f32x4;
typedef __attribute__((ext_vector_type(4)))  u32    u32x4;
typedef __attribute__((ext_vector_type(2)))  u32    u32x2;

#define ERA_N 40320
#define H_N   10242
#define LAT   256
#define E_E2H 120960
#define E_H2H 61452
#define E_H2E 120960

__device__ __forceinline__ u16 f2b(float f) {  // f32 -> bf16, round-to-nearest-even
  u32 u = __builtin_bit_cast(u32, f);
  u32 r = u + 0x7FFFu + ((u >> 16) & 1u);
  return (u16)(r >> 16);
}

union FragBF { u32x4 u[2]; v16bf v; };

// ---------------------------------------------------------------- tiny kernels
__global__ void zero_u16_k(u16* p, long n) {
  long i = (long)blockIdx.x * blockDim.x + threadIdx.x;
  if (i < n) p[i] = 0;
}
__global__ void zero_f32_k(float* p, long n) {
  long i = (long)blockIdx.x * blockDim.x + threadIdx.x;
  if (i < n) p[i] = 0.f;
}
__global__ void copy_f32_k(float* d, const float* s, long n) {
  long i = (long)blockIdx.x * blockDim.x + threadIdx.x;
  if (i < n) d[i] = s[i];
}
__global__ void add_f32_k(float* a, const float* b, long n) {
  long i = (long)blockIdx.x * blockDim.x + threadIdx.x;
  if (i < n) a[i] += b[i];
}
// zero-pad rows: dst[r, 0..kpad) = { src[r, 0..klen), 0... }
__global__ void pad_f32_k(const float* __restrict__ src, float* __restrict__ dst,
                          int klen, int kpad, long rows) {
  long i = (long)blockIdx.x * blockDim.x + threadIdx.x;
  long total = rows * kpad;
  if (i >= total) return;
  long r = i / kpad;
  int  c = (int)(i - r * kpad);
  dst[i] = (c < klen) ? src[r * klen + c] : 0.f;
}
// pack fp32 weight rows into bf16 COLUMN-MAJOR buffer: dst[n*Ktot + dstRow0+r] = src[(srcRow0+r)*N + n]
__global__ void pack_w_k(const float* __restrict__ src, u16* __restrict__ dst,
                         int rows, int N, int srcRow0, int dstRow0, int Ktot) {
  long i = (long)blockIdx.x * blockDim.x + threadIdx.x;
  long total = (long)rows * N;
  if (i >= total) return;
  int r = (int)(i / N), c = (int)(i - (long)r * N);
  dst[(long)c * Ktot + dstRow0 + r] = f2b(src[(long)(srcRow0 + r) * N + c]);
}
// segment-sum:  agg[dst[e], :] += msg[e, :]   (C = 256 hardcoded)
__global__ void scatter_add_k(const float* __restrict__ msg, const int* __restrict__ dst,
                              float* __restrict__ agg, long total) {
  long i = (long)blockIdx.x * blockDim.x + threadIdx.x;
  if (i >= total) return;
  long e = i >> 8;
  int  c = (int)(i & 255);
  unsafeAtomicAdd(&agg[(long)dst[e] * LAT + c], msg[i]);
}

// ---------------------------------------------------------------- WMMA GEMM
// out[M,N] = epilogue( concat_gathered_A[M, Ktot] @ W[Ktot, N] + bias )
// All A segments have stride == K extent (kpad, multiple of 32; ragged inputs are
// pre-padded on device). W is bf16 column-major [N][Ktot]: B fragments load straight
// from global (L2/WGP$-resident). A tile staged in double-buffered LDS, one quad per
// thread, branchless (row clamped to M-1; out-of-range rows never stored), with the
// next K-step's quad prefetched into registers across the WMMA.
// Block = 256 threads (8 waves): waves 0-3 own M rows 0-15, waves 4-7 rows 16-31;
// each wave owns TPW = N/64 N-tiles of 16.
template <int TPW>
__global__ void __launch_bounds__(256)
gemm_k(const float* a0, const int* i0, int kp0,
       const float* a1, const int* i1, int kp1,
       const float* a2, const int* i2, int kp2,
       const u16* __restrict__ W, int Ktot, const float* __restrict__ bias,
       int M, int N, int act,
       const float* __restrict__ lng, const float* __restrict__ lnb,
       const float* __restrict__ resid, int residLd, float* __restrict__ out)
{
  __shared__ __align__(16) u16 As[2][32 * 32];   // double-buffered A tile
  __shared__ float Cs[32 * 256];
  __shared__ float redS[256], redQ[256];
  __shared__ float mu_s[32], rs_s[32];

  const int tid   = threadIdx.x;
  const int lane  = tid & 31;
  const int wave  = tid >> 5;
  const int mhalf = wave >> 2;     // which 16-row half of the 32-row M tile
  const int wq    = wave & 3;      // N quadrant
  const int m0    = blockIdx.x * 32;

  const v8f vzero = {0.f, 0.f, 0.f, 0.f, 0.f, 0.f, 0.f, 0.f};
  v8f acc[TPW];
#pragma unroll
  for (int t = 0; t < TPW; ++t) acc[t] = vzero;

  // A staging: one 4-float quad per thread (256 threads = 32 rows x 8 quads).
  // Row clamped to M-1 -> branchless; clamped rows only feed discarded outputs.
  const int ar   = tid >> 3;
  const int akq  = (tid & 7) << 2;
  const int crow = min(m0 + ar, M - 1);
  const float* r0 = a0 ? a0 + (long)(i0 ? i0[crow] : crow) * kp0 + akq : nullptr;
  const float* r1 = a1 ? a1 + (long)(i1 ? i1[crow] : crow) * kp1 + akq : nullptr;
  const float* r2 = a2 ? a2 + (long)(i2 ? i2[crow] : crow) * kp2 + akq : nullptr;
  auto qptr = [&](int kc) -> const float* {
    if (kc < kp0) return r0 + kc;
    kc -= kp0;
    if (kc < kp1) return r1 + kc;
    return r2 + (kc - kp1);
  };

  // A fragment ds address (contiguous k order, matched by B):
  // lane<16 -> K{0..15}, lane>=16 -> K{16..31}
  const int afOff = (((mhalf << 4) + (lane & 15)) << 5) + ((lane >> 4) << 4);
  // B fragment: lane = column n0+(lane&15); lane>=16 -> K+16
  const u16* Wlane = W + (long)(lane & 15) * Ktot + ((lane >> 4) << 4);

  f32x4 q = *(const f32x4*)qptr(0);
  for (int kc = 0, p = 0; kc < Ktot; kc += 32, ++p) {
    u16* Asb = As[p & 1];
    // ---- pack current quad -> ds_store_b64
    u32x2 pk;
    pk.x = (u32)f2b(q.x) | ((u32)f2b(q.y) << 16);
    pk.y = (u32)f2b(q.z) | ((u32)f2b(q.w) << 16);
    *(u32x2*)(Asb + (ar << 5) + akq) = pk;
    __syncthreads();
    // ---- prefetch next K-step's quad (overlaps the WMMAs below)
    if (kc + 32 < Ktot) q = *(const f32x4*)qptr(kc + 32);
    // ---- A fragment from LDS
    FragBF af;
    af.u[0] = *(const u32x4*)(Asb + afOff);
    af.u[1] = *(const u32x4*)(Asb + afOff + 8);
    // ---- B fragments straight from global (column-major W) + WMMA
#pragma unroll
    for (int t = 0; t < TPW; ++t) {
      int n0 = (wq * TPW + t) << 4;
      const u16* Wc = Wlane + (long)n0 * Ktot + kc;
      FragBF bf_;
      bf_.u[0] = *(const u32x4*)(Wc);
      bf_.u[1] = *(const u32x4*)(Wc + 8);
      acc[t] = __builtin_amdgcn_wmma_f32_16x16x32_bf16(
          false, af.v, false, bf_.v, (short)0, acc[t], false, false);
    }
  }
  __syncthreads();

  // ---- dump accumulators to LDS (C layout: lane<16 -> M 0..7, lane>=16 -> M 8..15)
#pragma unroll
  for (int t = 0; t < TPW; ++t) {
    int n0 = (wq * TPW + t) << 4;
    int n  = n0 + (lane & 15);
    int mb = (mhalf << 4) + ((lane >> 4) << 3);
#pragma unroll
    for (int r = 0; r < 8; ++r) Cs[(mb + r) * N + n] = acc[t][r];
  }
  __syncthreads();
  // ---- bias + activation
  for (int i = tid; i < 32 * N; i += 256) {
    int n = i % N;
    float v = Cs[i] + bias[n];
    if (act == 1) v = v / (1.f + __expf(-v));  // SiLU
    Cs[i] = v;
  }
  __syncthreads();
  // ---- optional LayerNorm over each row (8 threads/row partials + combine)
  if (lng) {
    int rrow = tid >> 3, j = tid & 7, chunk = N >> 3;
    const float* rp = Cs + rrow * N + j * chunk;
    float s = 0.f, s2 = 0.f;
    for (int n = 0; n < chunk; ++n) { float v = rp[n]; s += v; s2 += v * v; }
    redS[tid] = s; redQ[tid] = s2;
    __syncthreads();
    if (tid < 32) {
      float ts = 0.f, tq = 0.f;
      for (int k = 0; k < 8; ++k) { ts += redS[tid * 8 + k]; tq += redQ[tid * 8 + k]; }
      float mu = ts / (float)N;
      mu_s[tid] = mu;
      rs_s[tid] = rsqrtf(tq / (float)N - mu * mu + 1e-5f);
    }
    __syncthreads();
  }
  // ---- normalize / residual / store
  for (int i = tid; i < 32 * N; i += 256) {
    int mm = i / N, n = i - mm * N;
    int row = m0 + mm;
    if (row >= M) continue;
    float v = Cs[i];
    if (lng)   v = (v - mu_s[mm]) * rs_s[mm] * lng[n] + lnb[n];
    if (resid) v += resid[(long)row * residLd + n];
    out[(long)row * N + n] = v;
  }
}

// ---------------------------------------------------------------- host side
struct Mlp2 { const float *w0, *w1, *b0, *b1, *g, *beta; };
struct MP   { Mlp2 ee; Mlp2 edge[4]; Mlp2 node[4]; int nb; };
struct PLin { const u16* w; const float* b; int kT; };
struct PMlp { PLin l0, l1; const float *g, *beta; };
struct PMP  { PMlp ee; PMlp edge[4]; PMlp node[4]; int nb; };
struct Seg  { const float* a; const int* idx; int kpad; };

static inline unsigned cdivu(long a, long b) { return (unsigned)((a + b - 1) / b); }

extern "C" void kernel_launch(void* const* d_in, const int* in_sizes, int n_in,
                              void* d_out, int out_size, void* d_ws, size_t ws_size,
                              hipStream_t stream) {
  (void)in_sizes; (void)n_in; (void)out_size; (void)ws_size;
  // ---- parse inputs (recursive dict-insertion order of setup_inputs())
  int cur = 0;
  auto F = [&]() { return (const float*)d_in[cur++]; };
  const float* x = F();  // [1, ERA, 80]
  auto takeMlp2 = [&]() { Mlp2 m; m.w0 = F(); m.w1 = F(); m.b0 = F(); m.b1 = F();
                          m.g = F(); m.beta = F(); return m; };
  Mlp2 era = takeMlp2();
  Mlp2 hem = takeMlp2();
  auto takeMP = [&](int nb) { MP mp; mp.nb = nb; mp.ee = takeMlp2();
    for (int i = 0; i < nb; ++i) { mp.edge[i] = takeMlp2(); mp.node[i] = takeMlp2(); }
    return mp; };
  MP fmap = takeMP(1), proc = takeMP(4), bmap = takeMP(1);
  const float *xw0 = F(), *xw1 = F(), *xw2 = F(), *xb0 = F(), *xb1 = F(), *xb2 = F();
  const float* era_latlons = F();
  const float* h_latlons   = F();
  const float* e2h_attr = F();
  const float* h2h_attr = F();
  const float* h2e_attr = F();
  const int* e2h = (const int*)d_in[cur++];
  const int* h2h = (const int*)d_in[cur++];
  const int* h2e = (const int*)d_in[cur++];

  // ---- workspace bump allocator (~560 MB)
  char* ws = (char*)d_ws;
  size_t off = 0;
  auto allocF = [&](size_t n) { off = (off + 255) & ~(size_t)255;
                                float* p = (float*)(ws + off); off += n * 4; return p; };
  const int MAXE = E_E2H;
  float* eBuf   = allocF((size_t)MAXE  * LAT);
  float* mBuf   = allocF((size_t)MAXE  * LAT);
  float* h1Buf  = allocF((size_t)MAXE  * LAT);
  float* aggBuf = allocF((size_t)ERA_N * LAT);
  float* xEra   = allocF((size_t)ERA_N * LAT);
  float* xH     = allocF((size_t)H_N   * LAT);
  float* xProc  = allocF((size_t)H_N   * LAT);
  // padded ragged inputs (so the GEMM K loop is branchless)
  float* xPad    = allocF((size_t)ERA_N * 96);
  float* eraLLP  = allocF((size_t)ERA_N * 32);
  float* hLLP    = allocF((size_t)H_N   * 32);
  float* e2hAttP = allocF((size_t)E_E2H * 32);
  float* h2hAttP = allocF((size_t)E_H2H * 32);
  float* h2eAttP = allocF((size_t)E_H2E * 32);
  auto pad = [&](const float* src, float* dst, int klen, int kpad, long rows) {
    pad_f32_k<<<cdivu(rows * kpad, 256), 256, 0, stream>>>(src, dst, klen, kpad, rows);
  };
  pad(x,           xPad,    80, 96, ERA_N);
  pad(era_latlons, eraLLP,  4,  32, ERA_N);
  pad(h_latlons,   hLLP,    4,  32, H_N);
  pad(e2h_attr,    e2hAttP, 3,  32, E_E2H);
  pad(h2h_attr,    h2hAttP, 3,  32, E_H2H);
  pad(h2e_attr,    h2eAttP, 3,  32, E_H2E);

  // ---- pack weights to zero-padded bf16 column-major [N][Ktot]
  auto packLin = [&](const float* w, const float* b, int N,
                     const int* klen, const int* kpad, int nseg) {
    int kT = 0; for (int i = 0; i < nseg; ++i) kT += kpad[i];
    off = (off + 255) & ~(size_t)255;
    u16* dst = (u16*)(ws + off); off += (size_t)kT * N * 2;
    long tot = (long)kT * N;
    zero_u16_k<<<cdivu(tot, 256), 256, 0, stream>>>(dst, tot);
    int sr = 0, dr = 0;
    for (int i = 0; i < nseg; ++i) {
      long t2 = (long)klen[i] * N;
      pack_w_k<<<cdivu(t2, 256), 256, 0, stream>>>(w, dst, klen[i], N, sr, dr, kT);
      sr += klen[i]; dr += kpad[i];
    }
    return PLin{dst, b, kT};
  };
  const int kl_era[2] = {80, 4}, kp_era[2] = {96, 32};
  const int kl3[1] = {3},   kp3[1] = {32};
  const int kl4[1] = {4},   kp4[1] = {32};
  const int kl256[1] = {256}, kp256[1] = {256};
  const int kl512[1] = {512}, kp512[1] = {512};
  const int kl768[1] = {768}, kp768[1] = {768};
  auto packMlp2 = [&](const Mlp2& m, const int* kl0, const int* kp0, int ns0) {
    PMlp P; P.l0 = packLin(m.w0, m.b0, 256, kl0, kp0, ns0);
    P.l1 = packLin(m.w1, m.b1, 256, kl256, kp256, 1);
    P.g = m.g; P.beta = m.beta; return P; };
  auto packMP = [&](const MP& mp) { PMP P; P.nb = mp.nb;
    P.ee = packMlp2(mp.ee, kl3, kp3, 1);
    for (int b = 0; b < mp.nb; ++b) {
      P.edge[b] = packMlp2(mp.edge[b], kl768, kp768, 1);
      P.node[b] = packMlp2(mp.node[b], kl512, kp512, 1);
    }
    return P; };
  PMlp Pera = packMlp2(era, kl_era, kp_era, 2);
  PMlp Ph   = packMlp2(hem, kl4, kp4, 1);
  PMP  Pf = packMP(fmap), Pp = packMP(proc), Pb = packMP(bmap);
  PLin ex0 = packLin(xw0, xb0, 256, kl256, kp256, 1);
  PLin ex1 = packLin(xw1, xb1, 256, kl256, kp256, 1);
  PLin ex2 = packLin(xw2, xb2, 64,  kl256, kp256, 1);

  // ---- launch helpers
  const Seg nil{nullptr, nullptr, 0};
  auto gemm = [&](Seg s0, Seg s1, Seg s2, PLin L, int M, int N, int act,
                  const float* g, const float* be, const float* resid, int residLd,
                  float* outp) {
    dim3 grid(cdivu(M, 32));
    if (N == 64)
      gemm_k<1><<<grid, 256, 0, stream>>>(
          s0.a, s0.idx, s0.kpad, s1.a, s1.idx, s1.kpad, s2.a, s2.idx, s2.kpad,
          L.w, L.kT, L.b, M, N, act, g, be, resid, residLd, outp);
    else
      gemm_k<4><<<grid, 256, 0, stream>>>(
          s0.a, s0.idx, s0.kpad, s1.a, s1.idx, s1.kpad, s2.a, s2.idx, s2.kpad,
          L.w, L.kT, L.b, M, N, act, g, be, resid, residLd, outp);
  };
  // interaction-network block sequence for one GNN (attrP = pre-padded [E,32] attrs)
  auto gnn = [&](const float* attrP, const int* src, const int* dst, int E,
                 const float* xs, float* xd, int nDst, const PMP& mp) {
    gemm(Seg{attrP, nullptr, 32}, nil, nil, mp.ee.l0, E, 256, 1, 0, 0, 0, 0, h1Buf);
    gemm(Seg{h1Buf, nullptr, 256}, nil, nil, mp.ee.l1, E, 256, 0,
         mp.ee.g, mp.ee.beta, 0, 0, eBuf);
    for (int b = 0; b < mp.nb; ++b) {
      // m = edgeMLP([xs[src], xd[dst], e])
      gemm(Seg{xs, src, 256}, Seg{xd, dst, 256}, Seg{eBuf, nullptr, 256},
           mp.edge[b].l0, E, 256, 1, 0, 0, 0, 0, h1Buf);
      gemm(Seg{h1Buf, nullptr, 256}, nil, nil, mp.edge[b].l1, E, 256, 0,
           mp.edge[b].g, mp.edge[b].beta, 0, 0, mBuf);
      long ne = (long)E * LAT;
      add_f32_k<<<cdivu(ne, 256), 256, 0, stream>>>(eBuf, mBuf, ne);   // e += m
      long na = (long)nDst * LAT;
      zero_f32_k<<<cdivu(na, 256), 256, 0, stream>>>(aggBuf, na);
      scatter_add_k<<<cdivu(ne, 256), 256, 0, stream>>>(mBuf, dst, aggBuf, ne);
      // xd += nodeMLP([xd, agg])   (in-place: residual fused into epilogue)
      gemm(Seg{xd, nullptr, 256}, Seg{aggBuf, nullptr, 256}, nil,
           mp.node[b].l0, nDst, 256, 1, 0, 0, 0, 0, h1Buf);
      gemm(Seg{h1Buf, nullptr, 256}, nil, nil, mp.node[b].l1, nDst, 256, 0,
           mp.node[b].g, mp.node[b].beta, xd, 256, xd);
    }
  };

  // ---- pipeline
  // era / h node embeddings
  gemm(Seg{xPad, nullptr, 96}, Seg{eraLLP, nullptr, 32}, nil,
       Pera.l0, ERA_N, 256, 1, 0, 0, 0, 0, h1Buf);
  gemm(Seg{h1Buf, nullptr, 256}, nil, nil, Pera.l1, ERA_N, 256, 0,
       Pera.g, Pera.beta, 0, 0, xEra);
  gemm(Seg{hLLP, nullptr, 32}, nil, nil, Ph.l0, H_N, 256, 1, 0, 0, 0, 0, h1Buf);
  gemm(Seg{h1Buf, nullptr, 256}, nil, nil, Ph.l1, H_N, 256, 0,
       Ph.g, Ph.beta, 0, 0, xH);
  // forward mapper era -> h (updates xH in place -> x_latent)
  gnn(e2hAttP, e2h, e2h + E_E2H, E_E2H, xEra, xH, H_N, Pf);
  // processor on h mesh + residual
  long nh = (long)H_N * LAT;
  copy_f32_k<<<cdivu(nh, 256), 256, 0, stream>>>(xProc, xH, nh);
  gnn(h2hAttP, h2h, h2h + E_H2H, E_H2H, xProc, xProc, H_N, Pp);  // homogeneous
  add_f32_k<<<cdivu(nh, 256), 256, 0, stream>>>(xProc, xH, nh);
  // backward mapper h -> era (updates xEra in place)
  gnn(h2eAttP, h2e, h2e + E_H2E, E_H2E, xProc, xEra, ERA_N, Pb);
  // extract MLP + input residual (x[..., :64]) fused into last GEMM
  gemm(Seg{xEra,  nullptr, 256}, nil, nil, ex0, ERA_N, 256, 1, 0, 0, 0, 0, h1Buf);
  gemm(Seg{h1Buf, nullptr, 256}, nil, nil, ex1, ERA_N, 256, 1, 0, 0, 0, 0, mBuf);
  gemm(Seg{mBuf,  nullptr, 256}, nil, nil, ex2, ERA_N, 64, 0, 0, 0,
       x, 80, (float*)d_out);
}
